// snn_conv_68221260530284
// MI455X (gfx1250) — matrix operand order repaired and checked by reading, compiled, not run on hardware
//
#include <hip/hip_runtime.h>
#include <stdint.h>

// y = sum_{k=0}^{4} L^k x W[k],  M=8192, F_IN=F_OUT=32, K=5.
// Memory-bound on L. Strategy: pass 1 streams fp32 L, converts to bf16 into
// workspace (128 MB -> fits MI455X 192 MB L2); passes 2-4 read the L2-resident
// bf16 copy. GEMM uses v_wmma_f32_16x16x32_bf16 with f32 accumulation.

#define MDIM 8192
#define FDIM 32
#define KCHE 5

typedef __attribute__((ext_vector_type(16))) __bf16   v16bf;
typedef __attribute__((ext_vector_type(8)))  float    v8f;
typedef __attribute__((ext_vector_type(4)))  uint32_t u32x4;
typedef __attribute__((ext_vector_type(4)))  float    f32x4;

union FragAB { u32x4 u[2]; v16bf v; };

__device__ __forceinline__ uint32_t f2bf1(float f) {
  uint32_t u = __builtin_bit_cast(uint32_t, f);
  return (u + 0x7FFFu + ((u >> 16) & 1u)) >> 16;   // round-to-nearest-even
}
__device__ __forceinline__ uint32_t pack2(float lo, float hi) {
  return f2bf1(lo) | (f2bf1(hi) << 16);
}

// ---------------------------------------------------------------------------
// z_next = L @ z_prev     (skinny GEMM: [8192x8192] @ [8192x32])
// Each workgroup (8 waves) owns a 16-row strip; waves split K (1024 each),
// partials reduced through LDS. MODE 0: read f32 L.  MODE 1: read f32 L and
// emit bf16 copy.  MODE 2: read bf16 L (L2-resident fast path).
// ---------------------------------------------------------------------------
template <int MODE>
__global__ __launch_bounds__(256)
void cheb_lz_gemm(const float* __restrict__ Lf,
                  const unsigned short* __restrict__ Lb,
                  unsigned short* __restrict__ LbOut,
                  const unsigned short* __restrict__ ZtIn,   // [32][8192] bf16
                  float* __restrict__ zOut,                  // [8192][32] f32
                  unsigned short* __restrict__ ZtOut)        // [32][8192] bf16
{
  __shared__ float red[8 * 16 * 32];          // 16 KB: per-wave partial tiles

  const int lane = threadIdx.x & 31;
  const int wave = threadIdx.x >> 5;
  const int r    = lane & 15;
  const int hi   = lane >> 4;                 // lane half (A/B layout split)
  const int rowBase = blockIdx.x * 16;
  const size_t arowOff = (size_t)(rowBase + r) * MDIM;
  const int kA = hi * 8;                      // A-frag: low regs K offset
  const int kB = hi * 8;                      // B-frag: K offset
  const int kc0 = wave * (MDIM / 8);          // 1024-wide K slice per wave

  v8f c0 = {};                                // N = 0..15
  v8f c1 = {};                                // N = 16..31

  for (int it = 0; it < (MDIM / 8) / 32; ++it) {
    const int kc = kc0 + it * 32;
    FragAB a, b0, b1;

    if (MODE == 2) {
      const unsigned short* ap = Lb + arowOff + kc + kA;
      a.u[0] = *(const u32x4*)(ap);           // K = kc+kA   .. +7
      a.u[1] = *(const u32x4*)(ap + 16);      // K = kc+kA+16.. +23
      __builtin_prefetch(ap + 256, 0, 0);
    } else {
      const float* ap = Lf + arowOff + kc + kA;
      f32x4 f0 = ((const f32x4*)ap)[0];
      f32x4 f1 = ((const f32x4*)ap)[1];
      f32x4 f2 = ((const f32x4*)ap)[4];       // ap + 16 floats
      f32x4 f3 = ((const f32x4*)ap)[5];
      u32x4 t0 = { pack2(f0.x, f0.y), pack2(f0.z, f0.w),
                   pack2(f1.x, f1.y), pack2(f1.z, f1.w) };
      u32x4 t1 = { pack2(f2.x, f2.y), pack2(f2.z, f2.w),
                   pack2(f3.x, f3.y), pack2(f3.z, f3.w) };
      a.u[0] = t0;
      a.u[1] = t1;
      __builtin_prefetch(ap + 256, 0, 0);
      if (MODE == 1) {                        // persist bf16 L (fits in L2)
        unsigned short* op = LbOut + arowOff + kc + kA;
        *(u32x4*)op        = a.u[0];
        *(u32x4*)(op + 16) = a.u[1];
      }
    }

    // B fragments: Zt row-contiguous in K -> two b128 loads per lane
    const unsigned short* bp0 = ZtIn + (size_t)r * MDIM + kc + kB;
    b0.u[0] = *(const u32x4*)bp0;
    b0.u[1] = *(const u32x4*)(bp0 + 16);
    const unsigned short* bp1 = ZtIn + (size_t)(r + 16) * MDIM + kc + kB;
    b1.u[0] = *(const u32x4*)bp1;
    b1.u[1] = *(const u32x4*)(bp1 + 16);

    c0 = __builtin_amdgcn_wmma_f32_16x16x32_bf16(false, a.v, false, b0.v,
                                                 (short)0, c0, false, false);
    c1 = __builtin_amdgcn_wmma_f32_16x16x32_bf16(false, a.v, false, b1.v,
                                                 (short)0, c1, false, false);
  }

  // Spill per-wave partial C tiles (16x32 f32) to LDS.
  // C layout: vgpr v -> row v + 8*hi, col = r (+16 for c1).
  float* myred = red + wave * 512;
  const int rowLoc = hi * 8;
  #pragma unroll
  for (int v = 0; v < 8; ++v) {
    myred[(rowLoc + v) * 32 + r]      = c0[v];
    myred[(rowLoc + v) * 32 + r + 16] = c1[v];
  }
  __syncthreads();

  // Cross-wave reduction: 512 outputs, 2 per thread. Write f32 z, stash
  // reduced tile in red[0..511] for the bf16 transpose pack.
  #pragma unroll
  for (int e = 0; e < 2; ++e) {
    const int o = threadIdx.x + e * 256;
    float s = 0.f;
    #pragma unroll
    for (int w = 0; w < 8; ++w) s += red[w * 512 + o];
    zOut[(size_t)(rowBase + (o >> 5)) * FDIM + (o & 31)] = s;
    red[o] = s;   // each entry o touched only by this thread in this phase
  }
  __syncthreads();

  // Transposed bf16 copy for next pass's B operand: col-major pack, b128 stores.
  if (threadIdx.x < 32) {
    const int cc = threadIdx.x;
    u32x4 p0 = { pack2(red[0 * 32 + cc],  red[1 * 32 + cc]),
                 pack2(red[2 * 32 + cc],  red[3 * 32 + cc]),
                 pack2(red[4 * 32 + cc],  red[5 * 32 + cc]),
                 pack2(red[6 * 32 + cc],  red[7 * 32 + cc]) };
    u32x4 p1 = { pack2(red[8 * 32 + cc],  red[9 * 32 + cc]),
                 pack2(red[10 * 32 + cc], red[11 * 32 + cc]),
                 pack2(red[12 * 32 + cc], red[13 * 32 + cc]),
                 pack2(red[14 * 32 + cc], red[15 * 32 + cc]) };
    unsigned short* zt = ZtOut + (size_t)cc * MDIM + rowBase;
    *(u32x4*)zt       = p0;
    *(u32x4*)(zt + 8) = p1;
  }
}

// x (f32 [8192][32]) -> Zt0 (bf16 [32][8192])
__global__ __launch_bounds__(256)
void x_to_zt(const float* __restrict__ x, unsigned short* __restrict__ Zt) {
  const int idx = blockIdx.x * 256 + threadIdx.x;
  const int row = idx >> 5, col = idx & 31;
  Zt[(size_t)col * MDIM + row] = (unsigned short)f2bf1(x[idx]);
}

// y (+)= z @ W[k]   -- tiny f32 GEMM, one output element per thread
__global__ __launch_bounds__(256)
void term_accum(const float* __restrict__ z, const float* __restrict__ Wk,
                float* __restrict__ y, const int accum) {
  const int idx = blockIdx.x * 256 + threadIdx.x;
  const int row = idx >> 5, col = idx & 31;
  float s = 0.f;
  #pragma unroll
  for (int i = 0; i < FDIM; ++i)
    s += z[(size_t)row * FDIM + i] * Wk[i * FDIM + col];
  if (accum) y[idx] += s; else y[idx] = s;
}

extern "C" void kernel_launch(void* const* d_in, const int* in_sizes, int n_in,
                              void* d_out, int out_size, void* d_ws, size_t ws_size,
                              hipStream_t stream) {
  (void)in_sizes; (void)n_in; (void)out_size;
  const float* x = (const float*)d_in[0];          // [8192][32]
  const float* L = (const float*)d_in[1];          // [8192][8192]
  const float* W = (const float*)d_in[2];          // [5][32][32]
  float* y = (float*)d_out;                        // [8192][32]

  const size_t LB = (size_t)MDIM * MDIM * sizeof(unsigned short);  // 128 MB
  const size_t ZT = (size_t)FDIM * MDIM * sizeof(unsigned short);  // 512 KB
  const size_t ZF = (size_t)MDIM * FDIM * sizeof(float);           // 1 MB
  const bool cacheLb = ws_size >= LB + 2 * ZT + ZF;

  char* ws = (char*)d_ws;
  unsigned short* Lb = (unsigned short*)ws;
  const size_t off = cacheLb ? LB : 0;
  unsigned short* ZtA = (unsigned short*)(ws + off);
  unsigned short* ZtB = (unsigned short*)(ws + off + ZT);
  float* zf = (float*)(ws + off + 2 * ZT);

  const int tblk = (MDIM * FDIM) / 256;            // 1024 blocks

  x_to_zt  <<<tblk, 256, 0, stream>>>(x, ZtA);     // z_0 (bf16, transposed)
  term_accum<<<tblk, 256, 0, stream>>>(x, W, y, 0);// y = x @ W[0]

  for (int k = 1; k < KCHE; ++k) {
    unsigned short* zin  = (k & 1) ? ZtA : ZtB;
    unsigned short* zout = (k & 1) ? ZtB : ZtA;
    if (cacheLb) {
      if (k == 1)
        cheb_lz_gemm<1><<<MDIM / 16, 256, 0, stream>>>(L, nullptr, Lb, zin, zf, zout);
      else
        cheb_lz_gemm<2><<<MDIM / 16, 256, 0, stream>>>(nullptr, Lb, nullptr, zin, zf, zout);
    } else {
      cheb_lz_gemm<0><<<MDIM / 16, 256, 0, stream>>>(L, nullptr, nullptr, zin, zf, zout);
    }
    term_accum<<<tblk, 256, 0, stream>>>(zf, W + k * FDIM * FDIM, y, 1);
  }
}
